// DegradationInteractionGraph_58875411694359
// MI455X (gfx1250) — matrix operand order repaired and checked by reading, not compiled, not run on hardware
//
#include <hip/hip_runtime.h>
#include <hip/hip_bf16.h>
#include <math.h>

// ---------------------------------------------------------------------------
// DegradationInteractionGraph for MI455X (gfx1250, wave32, WMMA bf16)
//   B=4, T=3, D=64, H=W=256.  Memory-bound: ~265MB @ 23.3TB/s (~11.4us floor)
//   -> all three GEMMs on v_wmma_f32_16x16x32_bf16, bf16 staging in LDS,
//      non-temporal global traffic so weights stay hot in WGP$/L2.
// ---------------------------------------------------------------------------

#define DD   64
#define TT   3
#define CH   192          // T*D
#define HW   65536        // 256*256
#define PIX  64           // pixels per workgroup (contiguous along W)
#define NTHREADS 256
#define NWAVES   8

typedef __attribute__((ext_vector_type(16))) __bf16 v16bf;
typedef __attribute__((ext_vector_type(8)))  float  v8f;
typedef __attribute__((ext_vector_type(4)))  float  fvec4;   // true clang vector
                                                             // (HIP float4 is a class)

// LDS strides (elements)
#define XP_STR  68        // u32 words per pair-row of x/cat (64 px + 4 pad -> b128 store ok)
#define W1_STR  72        // bf16 per row of W1^T   (64 + 8, multiple of 8 -> b128 ok)
#define WF_STR  208       // bf16 per row of Wf^T   (192 + 16, multiple of 8)
#define ACP_STR 70        // u32 words per pair-row of a/c (8*70 % 64 = 48 -> the 4
                          //  channel-groups of a pixel hit distinct bank sets)

union FragAB {
    v16bf v;
    struct { uint4 lo, hi; } q;
    unsigned int u[8];
};
union PackBF {
    __bf16 h[2];
    unsigned int u;
};

// A-fragment (16x32 bf16, row-major source W^T[e][d]):
//   lanes 0-15:  M=lane, VGPR0-3 = K 0..7,  VGPR4-7 = K 8..15
//   lanes 16-31: M=lane-16, K 16..23 / 24..31  (two ds_load_b128)
__device__ __forceinline__ FragAB load_a_frag(const __bf16* Wt, int stride,
                                              int m0, int k0, int lane) {
    FragAB f;
    const int m  = m0 + (lane & 15);
    const int kh = (lane < 16) ? 0 : 16;
    const __bf16* row = Wt + m * stride + k0 + kh;
    f.q.lo = *reinterpret_cast<const uint4*>(row);       // K kh..kh+7
    f.q.hi = *reinterpret_cast<const uint4*>(row + 8);   // K kh+8..kh+15
    return f;
}

// B-fragment (32x16 bf16) from pair-packed LDS (u32 = {K even lo, K odd hi}):
//   lanes 0-15:  N=lane,   K 0..15  (VGPR v = pair 2v,2v+1)
//   lanes 16-31: N=lane-16,K 16..31
__device__ __forceinline__ FragAB load_b_frag(const unsigned int* Xp,
                                              int k0, int p0, int lane) {
    FragAB f;
    const int n  = lane & 15;
    const int rp = (k0 + ((lane < 16) ? 0 : 16)) >> 1;
#pragma unroll
    for (int v = 0; v < 8; ++v)
        f.u[v] = Xp[(rp + v) * XP_STR + p0 + n];
    return f;
}

__global__ __launch_bounds__(NTHREADS)
void dig_fused_kernel(const float* __restrict__ xin,
                      const float* __restrict__ W1,
                      const float* __restrict__ b1,
                      const float* __restrict__ W2,
                      const float* __restrict__ b2,
                      const float* __restrict__ Wf,
                      const float* __restrict__ bf,
                      float* __restrict__ out)
{
    __shared__ unsigned int s_x  [(CH / 2) * XP_STR];    // bf16-pair packed x tile
    __shared__ unsigned int s_cat[(CH / 2) * XP_STR];    // bf16-pair packed cat tile
    __shared__ unsigned int s_apk[(CH / 2) * ACP_STR];   // bf16-pair packed a[t,e][p]
    __shared__ unsigned int s_cpk[(CH / 2) * ACP_STR];   // bf16-pair packed c[t,e][p]
    __shared__ __bf16 s_w1a[DD * W1_STR];                // W1[:D]^T  [e][d]
    __shared__ __bf16 s_w1c[DD * W1_STR];                // W1[D:]^T  [e][d]
    __shared__ __bf16 s_wf [DD * WF_STR];                // Wf^T      [e][cin]
    __shared__ float  s_w2[DD], s_b1[DD], s_bf[DD];
    __shared__ float  s_b2;

    const int tid  = threadIdx.x;
    const int lane = tid & 31;
    const int wid  = tid >> 5;

    // pixel tile: 64 contiguous pixels inside one row of one image
    const long long P0  = (long long)blockIdx.x * PIX;
    const int       bb  = (int)(P0 / HW);
    const int       rem = (int)(P0 % HW);

    // ---------------- phase 1: weights -> LDS (bf16, transposed) -----------
    // Regular-temporal loads: every block re-reads these ~81KB, so they stay
    // resident in WGP$/L2 (streaming x/out traffic below is non-temporal).
    for (int idx = tid; idx < 2 * DD * DD; idx += NTHREADS) {
        const int din = idx >> 6;            // 0..127
        const int e   = idx & 63;
        const float v = W1[idx];             // W1[din][e]
        if (din < DD) s_w1a[e * W1_STR + din]        = (__bf16)v;
        else          s_w1c[e * W1_STR + (din - DD)] = (__bf16)v;
    }
    for (int idx = tid; idx < CH * DD; idx += NTHREADS) {
        const int cin = idx >> 6;
        const int e   = idx & 63;
        s_wf[e * WF_STR + cin] = (__bf16)Wf[idx];    // Wf[cin][e]
    }
    if (tid < DD) {
        s_w2[tid] = W2[tid];
        s_b1[tid] = b1[tid];
        s_bf[tid] = bf[tid];
    }
    if (tid == 0) s_b2 = b2[0];

    // ---------------- phase 2: x tile -> LDS bf16 pair-packed --------------
    // Thread = (channel pair cp, 4-pixel group). Two NT global_load_b128,
    // pack 4 bf16 pairs, one ds_store_b128.
    {
        const int p4 = (tid & 15) * 4;
        for (int cp = tid >> 4; cp < CH / 2; cp += 16) {
            const fvec4* g0 = reinterpret_cast<const fvec4*>(
                xin + (long long)(bb * CH + 2 * cp) * HW + rem + p4);
            const fvec4* g1 = reinterpret_cast<const fvec4*>(
                xin + (long long)(bb * CH + 2 * cp + 1) * HW + rem + p4);
            const fvec4 x0 = __builtin_nontemporal_load(g0);
            const fvec4 x1 = __builtin_nontemporal_load(g1);
            PackBF k0, k1, k2, k3;
            k0.h[0] = (__bf16)x0.x; k0.h[1] = (__bf16)x1.x;
            k1.h[0] = (__bf16)x0.y; k1.h[1] = (__bf16)x1.y;
            k2.h[0] = (__bf16)x0.z; k2.h[1] = (__bf16)x1.z;
            k3.h[0] = (__bf16)x0.w; k3.h[1] = (__bf16)x1.w;
            uint4 w; w.x = k0.u; w.y = k1.u; w.z = k2.u; w.w = k3.u;
            *reinterpret_cast<uint4*>(&s_x[cp * XP_STR + p4]) = w;
        }
    }
    __syncthreads();

    // ---------------- phase 3: a = W1a^T x, c = W1c^T x  (WMMA) ------------
    // 96 tiles: {a,c} x t(3) x etile(4) x ptile(4); K = 64 -> 2 wmma each.
    // D rows per lane are consecutive e -> pack row pairs, 4x ds_store_b32.
    for (int tI = wid; tI < 96; tI += NWAVES) {
        const int pt  = tI & 3;
        const int et  = (tI >> 2) & 3;
        const int t   = (tI >> 4) % 3;
        const int mat = (tI >= 48);
        const __bf16* Wt = mat ? s_w1c : s_w1a;
        v8f acc = {};
#pragma unroll
        for (int ks = 0; ks < 2; ++ks) {
            FragAB A = load_a_frag(Wt, W1_STR, et * 16, ks * 32, lane);
            FragAB B = load_b_frag(s_x, t * DD + ks * 32, pt * 16, lane);
            acc = __builtin_amdgcn_wmma_f32_16x16x32_bf16(
                      false, A.v, false, B.v, (short)0, acc, false, false);
        }
        unsigned int* dst = mat ? s_cpk : s_apk;
        const int n  = lane & 15;
        const int mo = (lane >> 4) * 8;
        const int rb = (t * DD + et * 16 + mo) >> 1;   // pair-row base
#pragma unroll
        for (int k = 0; k < 4; ++k) {
            PackBF pk;
            pk.h[0] = (__bf16)acc[2 * k];
            pk.h[1] = (__bf16)acc[2 * k + 1];
            dst[(rb + k) * ACP_STR + pt * 16 + n] = pk.u;
        }
    }
    __syncthreads();

    // ---------------- phase 4: pairwise gating (VALU) ----------------------
    // thread = (pixel p, quarter g of the 64 channels); all LDS reads are
    // packed dwords (two bf16 channels per ds_load_b32).
    {
        const int p   = tid >> 2;
        const int g   = tid & 3;
        const int e0  = g * 16;
        const int pr0 = g * 8;            // pair-row offset inside a t-block

        float wgt[TT][TT];
#pragma unroll
        for (int i = 0; i < TT; ++i) {
#pragma unroll
            for (int j = 0; j < TT; ++j) {
                if (i == j) { wgt[i][j] = 0.f; continue; }
                float s = 0.f;
#pragma unroll
                for (int k = 0; k < 8; ++k) {
                    PackBF av, cv;
                    av.u = s_apk[(i * 32 + pr0 + k) * ACP_STR + p];
                    cv.u = s_cpk[(j * 32 + pr0 + k) * ACP_STR + p];
                    const float t0 = (float)av.h[0] + (float)cv.h[0] + s_b1[e0 + 2 * k];
                    const float t1 = (float)av.h[1] + (float)cv.h[1] + s_b1[e0 + 2 * k + 1];
                    const float h0 = 0.5f * t0 * (1.f + erff(t0 * 0.70710678118f));
                    const float h1 = 0.5f * t1 * (1.f + erff(t1 * 0.70710678118f));
                    s += h0 * s_w2[e0 + 2 * k] + h1 * s_w2[e0 + 2 * k + 1];
                }
                // reduce partial dot across the 4 lanes of this pixel
                s += __shfl_xor(s, 1, 32);
                s += __shfl_xor(s, 2, 32);
                wgt[i][j] = 1.f / (1.f + expf(-(s + s_b2)));
            }
        }

#pragma unroll
        for (int i = 0; i < TT; ++i) {
            float acc[16];
#pragma unroll
            for (int k = 0; k < 8; ++k) {
                PackBF xv; xv.u = s_x[(i * 32 + pr0 + k) * XP_STR + p];
                acc[2 * k]     = (float)xv.h[0];
                acc[2 * k + 1] = (float)xv.h[1];
            }
#pragma unroll
            for (int j = 0; j < TT; ++j) {
                if (j == i) continue;
                const float w = wgt[i][j];
#pragma unroll
                for (int k = 0; k < 8; ++k) {
                    PackBF xv; xv.u = s_x[(j * 32 + pr0 + k) * XP_STR + p];
                    acc[2 * k]     += w * (float)xv.h[0];
                    acc[2 * k + 1] += w * (float)xv.h[1];
                }
            }
#pragma unroll
            for (int k = 0; k < 8; ++k) {
                PackBF pk;
                pk.h[0] = (__bf16)acc[2 * k];
                pk.h[1] = (__bf16)acc[2 * k + 1];
                s_cat[(i * 32 + pr0 + k) * XP_STR + p] = pk.u;
            }
        }
    }
    __syncthreads();

    // ---------------- phase 5: out = Wf^T cat + bf  (WMMA) -----------------
    for (int tI = wid; tI < 16; tI += NWAVES) {
        const int pt = tI & 3;
        const int et = tI >> 2;
        v8f acc = {};
#pragma unroll
        for (int ks = 0; ks < 6; ++ks) {
            FragAB A = load_a_frag(s_wf, WF_STR, et * 16, ks * 32, lane);
            FragAB B = load_b_frag(s_cat, ks * 32, pt * 16, lane);
            acc = __builtin_amdgcn_wmma_f32_16x16x32_bf16(
                      false, A.v, false, B.v, (short)0, acc, false, false);
        }
        const int n  = lane & 15;
        const int mo = (lane >> 4) * 8;
#pragma unroll
        for (int r = 0; r < 8; ++r) {
            const int e = et * 16 + mo + r;
            float* dst = out + (long long)(bb * DD + e) * HW + rem + pt * 16 + n;
            __builtin_nontemporal_store(acc[r] + s_bf[e], dst);
        }
    }
}

extern "C" void kernel_launch(void* const* d_in, const int* in_sizes, int n_in,
                              void* d_out, int out_size, void* d_ws, size_t ws_size,
                              hipStream_t stream) {
    const float* xin = (const float*)d_in[0];   // (B, 192, 256, 256)
    const float* W1  = (const float*)d_in[1];   // (128, 64)
    const float* b1  = (const float*)d_in[2];   // (64,)
    const float* W2  = (const float*)d_in[3];   // (64, 1)
    const float* b2  = (const float*)d_in[4];   // (1,)
    const float* Wf  = (const float*)d_in[5];   // (192, 64)
    const float* bf  = (const float*)d_in[6];   // (64,)
    float* out = (float*)d_out;                 // (B, 64, 256, 256)

    const int B = in_sizes[0] / (CH * HW);      // 4
    const int nblocks = (B * HW) / PIX;         // 4096
    dig_fused_kernel<<<nblocks, NTHREADS, 0, stream>>>(xin, W1, b1, W2, b2, Wf, bf, out);
}
